// PointNet2MSG_DepthContrast_39333310496887
// MI455X (gfx1250) — matrix operand order repaired and checked by reading, not compiled, not run on hardware
//
#include <hip/hip_runtime.h>
#include <hip/hip_bf16.h>
#include <cstdint>
#include <cstddef>

typedef _Float16 h16;
typedef __attribute__((ext_vector_type(16))) _Float16 v16h;
typedef __attribute__((ext_vector_type(8)))  _Float16 v8h;
typedef __attribute__((ext_vector_type(8)))  float    v8f;

#define BATCH 4
#define GAS __attribute__((address_space(1)))

// Force global (addrspace 1) vector loads/stores: avoids flat_* lowering, which
// would tick DScnt and occupy the LDS request path.
__device__ __forceinline__ v16h ldg_v16h(const h16* p) {
  return *(const GAS v16h*)(const GAS h16*)p;
}
__device__ __forceinline__ void stg_v8h(h16* p, v8h v) {
  *(GAS v8h*)(GAS h16*)p = v;
}

__host__ __device__ __forceinline__ int pad32i(int x) { return (x + 31) & ~31; }

// Swizzled activation layout (B-fragment native):
// element (k, col):
//   tile = col>>4, chunk = k>>5, lane = (col&15) + 16*((k>>4)&1), elem = k&15
//   addr = tile*(Cp*16) + chunk*512 + lane*16 + elem     [halves]
__device__ __forceinline__ size_t swzaddr(int k, int col, int Cp) {
  return (size_t)(col >> 4) * ((size_t)Cp * 16) + (size_t)(k >> 5) * 512
       + (size_t)((col & 15) + (((k >> 4) & 1) << 4)) * 16 + (size_t)(k & 15);
}

// ---------------------------------------------------------------- prep
__global__ void prep_kernel(const float* __restrict__ pc, float* __restrict__ xyz,
                            float* __restrict__ feats, int N) {
  int t = blockIdx.x * blockDim.x + threadIdx.x;
  if (t >= BATCH * N) return;
  float4 p = ((const float4*)pc)[t];
  xyz[(size_t)t * 3 + 0] = p.x;
  xyz[(size_t)t * 3 + 1] = p.y;
  xyz[(size_t)t * 3 + 2] = p.z;
  feats[t] = p.w;
}

// ---------------------------------------------------------------- FPS
__global__ void fps_kernel(const float* __restrict__ xyz, float* __restrict__ dist,
                           int* __restrict__ fidx, int N, int npoint) {
  int b = blockIdx.x;
  const float* X = xyz + (size_t)b * N * 3;
  float* ds = dist + (size_t)b * N;
  int* out = fidx + (size_t)b * npoint;
  __shared__ float sv[8];
  __shared__ int   si[8];
  __shared__ int   sfar;
  int t = threadIdx.x;
  int lane = t & 31, w = t >> 5;
  for (int i = t; i < N; i += blockDim.x) ds[i] = 1e10f;
  if (t == 0) sfar = 0;
  __syncthreads();
  for (int it = 0; it < npoint; ++it) {
    int far = sfar;
    if (t == 0) out[it] = far;
    float cx = X[(size_t)far * 3], cy = X[(size_t)far * 3 + 1], cz = X[(size_t)far * 3 + 2];
    float bestv = -1.0f; int besti = 0;
    for (int i = t; i < N; i += blockDim.x) {
      float dx = X[(size_t)i * 3] - cx;
      float dy = X[(size_t)i * 3 + 1] - cy;
      float dz = X[(size_t)i * 3 + 2] - cz;
      float d = dx * dx + dy * dy + dz * dz;
      float nd = fminf(ds[i], d);
      ds[i] = nd;
      if (nd > bestv) { bestv = nd; besti = i; }
    }
    // wave32 shuffle reduction (max value, tie -> smaller index)
#pragma unroll
    for (int off = 16; off > 0; off >>= 1) {
      float ov = __shfl_xor(bestv, off, 32);
      int   oi = __shfl_xor(besti, off, 32);
      if (ov > bestv || (ov == bestv && oi < besti)) { bestv = ov; besti = oi; }
    }
    if (lane == 0) { sv[w] = bestv; si[w] = besti; }
    __syncthreads();
    if (w == 0) {
      float v = (lane < 8) ? sv[lane] : -2.0f;
      int   ii = (lane < 8) ? si[lane] : 0;
#pragma unroll
      for (int off = 4; off > 0; off >>= 1) {
        float ov = __shfl_xor(v, off, 32);
        int   oi = __shfl_xor(ii, off, 32);
        if (ov > v || (ov == v && oi < ii)) { v = ov; ii = oi; }
      }
      if (lane == 0) sfar = ii;
    }
    __syncthreads();
  }
}

__global__ void gather_xyz_kernel(const float* __restrict__ xyz, const int* __restrict__ fidx,
                                  float* __restrict__ nxyz, int S, int N) {
  int t = blockIdx.x * blockDim.x + threadIdx.x;
  if (t >= BATCH * S) return;
  int b = t / S, s = t - b * S;
  int j = fidx[(size_t)b * S + s];
  const float* src = xyz + ((size_t)b * N + j) * 3;
  float* dst = nxyz + (size_t)t * 3;
  dst[0] = src[0]; dst[1] = src[1]; dst[2] = src[2];
}

// ---------------------------------------------------------------- ball query (two radii, one pass)
__global__ void ball_query_kernel(const float* __restrict__ nxyz, const float* __restrict__ xyz,
                                  int* __restrict__ idx0, int* __restrict__ idx1,
                                  int S, int N, float r0sq, float r1sq, int K0, int K1) {
  int t = blockIdx.x * blockDim.x + threadIdx.x;
  if (t >= BATCH * S) return;
  int b = t / S;
  const float* X = xyz + (size_t)b * N * 3;
  float cx = nxyz[(size_t)t * 3], cy = nxyz[(size_t)t * 3 + 1], cz = nxyz[(size_t)t * 3 + 2];
  int* o0 = idx0 + (size_t)t * K0;
  int* o1 = idx1 + (size_t)t * K1;
  int c0 = 0, c1 = 0, f0 = N - 1, f1 = N - 1;
  for (int n = 0; n < N; ++n) {
    float dx = X[(size_t)n * 3] - cx;
    float dy = X[(size_t)n * 3 + 1] - cy;
    float dz = X[(size_t)n * 3 + 2] - cz;
    float d2 = dx * dx + dy * dy + dz * dz;
    if (d2 < r1sq) {
      if (c1 == 0) f1 = n;
      if (c1 < K1) o1[c1++] = n;
      if (d2 < r0sq) {
        if (c0 == 0) f0 = n;
        if (c0 < K0) o0[c0++] = n;
      }
      if (c1 >= K1 && c0 >= K0) break;
    }
  }
  for (int k = c0; k < K0; ++k) o0[k] = f0;
  for (int k = c1; k < K1; ++k) o1[k] = f1;
}

// ---------------------------------------------------------------- grouping -> swizzled X
__global__ void group_kernel(const float* __restrict__ xyz, const float* __restrict__ nxyz,
                             const float* __restrict__ feats, const int* __restrict__ idx,
                             h16* __restrict__ Xst, int S, int K, int N, int Cf, int Cp) {
  int col = blockIdx.x * blockDim.x + threadIdx.x;
  int cols = S * K;
  if (col >= cols) return;
  int s = col / K;
  int j = idx[col];
  if (j >= N) j = N - 1;
  float gx = xyz[(size_t)j * 3]     - nxyz[(size_t)s * 3];
  float gy = xyz[(size_t)j * 3 + 1] - nxyz[(size_t)s * 3 + 1];
  float gz = xyz[(size_t)j * 3 + 2] - nxyz[(size_t)s * 3 + 2];
  Xst[swzaddr(0, col, Cp)] = (h16)gx;
  Xst[swzaddr(1, col, Cp)] = (h16)gy;
  Xst[swzaddr(2, col, Cp)] = (h16)gz;
  for (int c = 0; c < Cf; ++c)
    Xst[swzaddr(3 + c, col, Cp)] = (h16)feats[(size_t)c * N + j];
  for (int c = Cf + 3; c < Cp; ++c)
    Xst[swzaddr(c, col, Cp)] = (h16)0.0f;
}

// ---------------------------------------------------------------- weight pre-swizzle (A-fragment layout)
__global__ void swz_weights_kernel(const float* __restrict__ W, h16* __restrict__ Ws,
                                   int O, int C, int Mt, int kcTot) {
  int t = blockIdx.x * blockDim.x + threadIdx.x;
  int tot = Mt * kcTot * 32;
  if (t >= tot) return;
  int l  = t & 31;
  int kc = (t >> 5) % kcTot;
  int tm = t / (kcTot * 32);
  int m = l & 15, hh = l >> 4;
  int row = tm * 16 + m;
  h16* out = Ws + ((size_t)(tm * kcTot + kc) * 512 + (size_t)l * 16);
#pragma unroll
  for (int i = 0; i < 16; ++i) {
    int kk = hh * 16 + i;
    int k = kc * 32 + kk;
    float v = (row < O && k < C) ? W[(size_t)row * C + k] : 0.0f;
    out[i] = (h16)v;
  }
}

// ---------------------------------------------------------------- WMMA GEMM + bias + ReLU
// NT column tiles per wave: one A-fragment load feeds NT independent WMMA chains.
// Branch-free K loop; all fragment traffic forced to global address space.
template <int WAVES, int NT>
__global__ void wmma_gemm_swz(const h16* __restrict__ Ws, const float* __restrict__ bias,
                              const h16* __restrict__ X, h16* __restrict__ D,
                              int O, int Cp, int Ncols, int Dp) {
  int lane = threadIdx.x & 31;
  int wave = threadIdx.x >> 5;
  int tiles = Ncols >> 4;
  int tile0 = (blockIdx.x * WAVES + wave) * NT;
  if (tile0 >= tiles) return;          // wave-uniform
  int tileM = blockIdx.y;
  int kcTot = Cp >> 5;
  const h16* Ap = Ws + (size_t)tileM * kcTot * 512 + (size_t)lane * 16;
  const h16* Bp[NT];
  int tn[NT];
#pragma unroll
  for (int q = 0; q < NT; ++q) {
    int tq = tile0 + q;
    tn[q] = (tq < tiles) ? tq : (tiles - 1);   // clamp: duplicate tiles write identical data
    Bp[q] = X + (size_t)tn[q] * Cp * 16 + (size_t)lane * 16;
  }
  v8f acc[NT];
#pragma unroll
  for (int q = 0; q < NT; ++q) acc[q] = v8f{};
  for (int kc = 0; kc < kcTot; ++kc) {
    v16h a = ldg_v16h(Ap);
    Ap += 512;
#pragma unroll
    for (int q = 0; q < NT; ++q) {
      v16h bb = ldg_v16h(Bp[q]);
      __builtin_prefetch(Bp[q] + 1024, 0, 1);   // speculative, 2 chunks ahead
      acc[q] = __builtin_amdgcn_wmma_f32_16x16x32_f16(false, a, false, bb, (short)0,
                                                      acc[q], false, false);
      Bp[q] += 512;
    }
  }
  int n = lane & 15, hh = lane >> 4;
  int robase = tileM * 16 + hh * 8;
  float bv[8];
  if (robase + 8 <= O) {
#pragma unroll
    for (int r = 0; r < 8; ++r) bv[r] = bias[robase + r];
  } else {
#pragma unroll
    for (int r = 0; r < 8; ++r) bv[r] = (robase + r < O) ? bias[robase + r] : 0.0f;
  }
#pragma unroll
  for (int q = 0; q < NT; ++q) {
    v8h outv;
#pragma unroll
    for (int r = 0; r < 8; ++r)
      outv[r] = (h16)fmaxf(acc[q][r] + bv[r], 0.0f);
    h16* dp = D + (size_t)tn[q] * Dp * 16 + (size_t)(tileM >> 1) * 512
                + (size_t)(n + ((tileM & 1) << 4)) * 16 + (size_t)hh * 8;
    stg_v8h(dp, outv);
  }
}

// ---------------------------------------------------------------- max over K neighbors (swizzled in)
__global__ void maxpool_kernel(const h16* __restrict__ G, float* __restrict__ out,
                               int O, int S, int K, int Dp) {
  int t = blockIdx.x * blockDim.x + threadIdx.x;
  if (t >= O * S) return;
  int o = t / S, s = t - o * S;
  float m = -1e30f;
  for (int k = 0; k < K; ++k) m = fmaxf(m, (float)G[swzaddr(o, s * K + k, Dp)]);
  out[(size_t)o * S + s] = m;
}

// ---------------------------------------------------------------- FP: fused 3-NN + interp + concat
__global__ void fp_input_kernel(const float* __restrict__ uxyz, const float* __restrict__ kxyz,
                                const float* __restrict__ kfeat, const float* __restrict__ ufeat,
                                h16* __restrict__ Xst, int Nu, int Nk, int C1, int C2, int Cp) {
  int n = blockIdx.x * blockDim.x + threadIdx.x;
  if (n >= Nu) return;
  float px = uxyz[(size_t)n * 3], py = uxyz[(size_t)n * 3 + 1], pz = uxyz[(size_t)n * 3 + 2];
  float d0 = 1e30f, d1 = 1e30f, d2v = 1e30f;
  int i0 = 0, i1 = 0, i2 = 0;
  for (int j = 0; j < Nk; ++j) {
    float dx = kxyz[(size_t)j * 3] - px;
    float dy = kxyz[(size_t)j * 3 + 1] - py;
    float dz = kxyz[(size_t)j * 3 + 2] - pz;
    float d = dx * dx + dy * dy + dz * dz;
    if (d < d0)       { d2v = d1; i2 = i1; d1 = d0; i1 = i0; d0 = d; i0 = j; }
    else if (d < d1)  { d2v = d1; i2 = i1; d1 = d;  i1 = j; }
    else if (d < d2v) { d2v = d;  i2 = j; }
  }
  float w0 = 1.0f / (d0 + 1e-8f), w1 = 1.0f / (d1 + 1e-8f), w2 = 1.0f / (d2v + 1e-8f);
  float ws = w0 + w1 + w2;
  w0 /= ws; w1 /= ws; w2 /= ws;
  for (int c = 0; c < C1; ++c) {
    const float* f = kfeat + (size_t)c * Nk;
    Xst[swzaddr(c, n, Cp)] = (h16)(w0 * f[i0] + w1 * f[i1] + w2 * f[i2]);
  }
  for (int c = 0; c < C2; ++c)
    Xst[swzaddr(C1 + c, n, Cp)] = (h16)ufeat[(size_t)c * Nu + n];
  for (int c = C1 + C2; c < Cp; ++c)
    Xst[swzaddr(c, n, Cp)] = (h16)0.0f;
}

__global__ void copy_h2f_kernel(const h16* __restrict__ src, float* __restrict__ dst,
                                int O, int Nu, int Dp) {
  int t = blockIdx.x * blockDim.x + threadIdx.x;
  if (t >= O * Nu) return;
  int o = t / Nu, n = t - o * Nu;
  dst[t] = (float)src[swzaddr(o, n, Dp)];
}

__global__ void final_max_kernel(const h16* __restrict__ G, float* __restrict__ out,
                                 int Ch, int N, int Dp) {
  int c = threadIdx.x;
  if (c >= Ch) return;
  float m = -1e30f;
  for (int n = 0; n < N; ++n) m = fmaxf(m, (float)G[swzaddr(c, n, Dp)]);
  out[c] = m;
}

// ---------------------------------------------------------------- host orchestration
extern "C" void kernel_launch(void* const* d_in, const int* in_sizes, int n_in,
                              void* d_out, int out_size, void* d_ws, size_t ws_size,
                              hipStream_t stream) {
  (void)in_sizes; (void)n_in; (void)out_size; (void)ws_size;
  const float* pc = (const float*)d_in[0];
  // params flattened jax-tree order: 'fp' before 'sa', 'b' before 'w'
  auto fpB = [&](int k, int l) { return (const float*)d_in[1 + k * 4 + l * 2]; };
  auto fpW = [&](int k, int l) { return (const float*)d_in[2 + k * 4 + l * 2]; };
  auto saB = [&](int k, int sc, int l) { return (const float*)d_in[17 + ((k * 2 + sc) * 3 + l) * 2]; };
  auto saW = [&](int k, int sc, int l) { return (const float*)d_in[18 + ((k * 2 + sc) * 3 + l) * 2]; };

  static const int Ns[5] = {16384, 4096, 1024, 256, 64};
  static const int Cs[5] = {1, 96, 256, 512, 1024};
  static const float Rad[4][2] = {{0.1f, 0.5f}, {0.5f, 1.0f}, {1.0f, 2.0f}, {2.0f, 4.0f}};
  static const int Ksamp[2] = {16, 32};
  static const int SAD[4][2][4] = {
      {{4, 16, 16, 32}, {4, 32, 32, 64}},
      {{99, 64, 64, 128}, {99, 64, 96, 128}},
      {{259, 128, 196, 256}, {259, 128, 196, 256}},
      {{515, 256, 256, 512}, {515, 256, 384, 512}}};
  static const int FPD[4][3] = {{257, 128, 128}, {608, 256, 256}, {768, 512, 512}, {1536, 512, 512}};

  const int WV = 4, NT = 4, TPB = WV * NT;  // column tiles per block

  uintptr_t base = (uintptr_t)d_ws;
  size_t off = 0;
  auto alloc = [&](size_t bytes) -> void* {
    void* r = (void*)(base + off);
    off = (off + bytes + 255) & ~(size_t)255;
    return r;
  };
  float* xyzL[5]; float* featL[5];
  for (int k = 0; k < 5; ++k) xyzL[k] = (float*)alloc((size_t)BATCH * Ns[k] * 3 * 4);
  for (int k = 0; k < 5; ++k) featL[k] = (float*)alloc((size_t)BATCH * Cs[k] * Ns[k] * 4);
  float* nf3 = (float*)alloc((size_t)BATCH * 512 * 256 * 4);
  float* nf2 = (float*)alloc((size_t)BATCH * 512 * 1024 * 4);
  float* nf1 = (float*)alloc((size_t)BATCH * 256 * 4096 * 4);
  float* dist = (float*)alloc((size_t)BATCH * 16384 * 4);
  int* fidx = (int*)alloc((size_t)BATCH * 4096 * 4);
  int* idx16 = (int*)alloc((size_t)BATCH * 4096 * 16 * 4);
  int* idx32 = (int*)alloc((size_t)BATCH * 4096 * 32 * 4);
  h16* Xst = (h16*)alloc((size_t)10 * 1024 * 1024);
  h16* G0 = (h16*)alloc((size_t)17 * 1024 * 1024);
  h16* G1 = (h16*)alloc((size_t)17 * 1024 * 1024);
  h16* Wsw[3];
  for (int i = 0; i < 3; ++i) Wsw[i] = (h16*)alloc((size_t)2 * 1024 * 1024);

  // prep
  {
    int tot = BATCH * Ns[0];
    prep_kernel<<<(tot + 255) / 256, 256, 0, stream>>>(pc, xyzL[0], featL[0], Ns[0]);
  }

  // ---------------- SA layers ----------------
  for (int k = 0; k < 4; ++k) {
    int N = Ns[k], S = Ns[k + 1];
    fps_kernel<<<BATCH, 256, 0, stream>>>(xyzL[k], dist, fidx, N, S);
    {
      int tot = BATCH * S;
      gather_xyz_kernel<<<(tot + 255) / 256, 256, 0, stream>>>(xyzL[k], fidx, xyzL[k + 1], S, N);
    }
    {
      int tot = BATCH * S;
      float r0 = Rad[k][0], r1 = Rad[k][1];
      ball_query_kernel<<<(tot + 127) / 128, 128, 0, stream>>>(
          xyzL[k + 1], xyzL[k], idx16, idx32, S, N, r0 * r0, r1 * r1, 16, 32);
    }
    int coff = 0;
    for (int sc = 0; sc < 2; ++sc) {
      int K = Ksamp[sc], cols = S * K;
      const int* dims = SAD[k][sc];
      int Cp0 = pad32i(dims[0]);
      // pre-swizzle the 3 weight matrices (once per scale, reused across batches)
      {
        int Cc = dims[0];
        for (int l = 0; l < 3; ++l) {
          int O = dims[l + 1];
          int Mt = pad32i(O) / 16, kcT = pad32i(Cc) / 32;
          int tot = Mt * kcT * 32;
          swz_weights_kernel<<<(tot + 255) / 256, 256, 0, stream>>>(
              saW(k, sc, l), Wsw[l], O, Cc, Mt, kcT);
          Cc = O;
        }
      }
      for (int b = 0; b < BATCH; ++b) {
        group_kernel<<<(cols + 255) / 256, 256, 0, stream>>>(
            xyzL[k] + (size_t)b * N * 3, xyzL[k + 1] + (size_t)b * S * 3,
            featL[k] + (size_t)b * Cs[k] * N,
            (sc ? idx32 : idx16) + (size_t)b * S * K,
            Xst, S, K, N, Cs[k], Cp0);
        const h16* cur = Xst;
        int Cc = dims[0];
        h16* bufs[3] = {G0, G1, G0};
        for (int l = 0; l < 3; ++l) {
          int O = dims[l + 1];
          int Cp = pad32i(Cc), Dp = pad32i(O);
          int tiles = cols / 16;
          dim3 grid((tiles + TPB - 1) / TPB, Dp / 16, 1);
          wmma_gemm_swz<WV, NT><<<grid, WV * 32, 0, stream>>>(
              Wsw[l], saB(k, sc, l), cur, bufs[l], O, Cp, cols, Dp);
          cur = bufs[l]; Cc = O;
        }
        int O = dims[3];
        int tot = O * S;
        maxpool_kernel<<<(tot + 255) / 256, 256, 0, stream>>>(
            cur, featL[k + 1] + ((size_t)b * Cs[k + 1] + coff) * S, O, S, K, pad32i(O));
      }
      coff += dims[3];
    }
  }

  // ---------------- FP layers (deepest first) ----------------
  const float* knF[4] = {nf1, nf2, nf3, featL[4]};
  float* outF[4] = {nullptr, nf1, nf2, nf3};
  const int C1s[4] = {256, 512, 512, 1024};
  const int C2s[4] = {1, 96, 256, 512};
  for (int t = 3; t >= 0; --t) {
    int Nu = Ns[t], Nk = Ns[t + 1];
    int C1 = C1s[t], C2 = C2s[t];
    const int* dims = FPD[t];
    int Cp0 = pad32i(dims[0]);
    {
      int Cc = dims[0];
      for (int l = 0; l < 2; ++l) {
        int O = dims[l + 1];
        int Mt = pad32i(O) / 16, kcT = pad32i(Cc) / 32;
        int tot = Mt * kcT * 32;
        swz_weights_kernel<<<(tot + 255) / 256, 256, 0, stream>>>(
            fpW(t, l), Wsw[l], O, Cc, Mt, kcT);
        Cc = O;
      }
    }
    for (int b = 0; b < BATCH; ++b) {
      fp_input_kernel<<<(Nu + 127) / 128, 128, 0, stream>>>(
          xyzL[t] + (size_t)b * Nu * 3, xyzL[t + 1] + (size_t)b * Nk * 3,
          knF[t] + (size_t)b * C1 * Nk, featL[t] + (size_t)b * C2 * Nu,
          Xst, Nu, Nk, C1, C2, Cp0);
      const h16* cur = Xst;
      int Cc = dims[0];
      h16* bufs[2] = {G0, G1};
      for (int l = 0; l < 2; ++l) {
        int O = dims[l + 1];
        int Cp = pad32i(Cc), Dp = pad32i(O);
        int tiles = Nu / 16;
        dim3 grid((tiles + TPB - 1) / TPB, Dp / 16, 1);
        wmma_gemm_swz<WV, NT><<<grid, WV * 32, 0, stream>>>(
            Wsw[l], fpB(t, l), cur, bufs[l], O, Cp, Nu, Dp);
        cur = bufs[l]; Cc = O;
      }
      int O = dims[2];
      if (t > 0) {
        int tot = O * Nu;
        copy_h2f_kernel<<<(tot + 255) / 256, 256, 0, stream>>>(
            cur, outF[t] + (size_t)b * O * Nu, O, Nu, pad32i(O));
      } else {
        final_max_kernel<<<1, 128, 0, stream>>>(cur, (float*)d_out + (size_t)b * 128, O, Nu,
                                                pad32i(O));
      }
    }
  }
}
